// DeltaGRUCell_87488483819953
// MI455X (gfx1250) — matrix operand order repaired and checked by reading, not compiled, not run on hardware
//
#include <hip/hip_runtime.h>

typedef __attribute__((ext_vector_type(16))) _Float16 v16h;
typedef __attribute__((ext_vector_type(8)))  float    v8f;

#define SEQ    2048
#define NBATCH 32
#define NI     256
#define NH     512
#define H3     1536
#define NBLK   32
#define KT_I   8     /* 256/32 */
#define KT_H   16    /* 512/32 */
#define THRESH 0.25f

// ---------------- quantization helpers (Q m.n fixed point, STE round) -------
__device__ __forceinline__ float quantf(float x, float power, float invp, float clip) {
    float v = rintf(x * power);
    v = fminf(fmaxf(v, -clip), clip - 1.0f);
    return v * invp;
}
__device__ __forceinline__ float sigm(float x) { return 1.0f / (1.0f + expf(-x)); }

// ---------------- WMMA f16 A-fragment (16x32 MxK) element address -----------
// lanes 0-15: M=0..15, K in {0..7,16..23}; lanes 16-31: M=0..15, K in {8..15,24..31}
// VGPR v (v16h elements 2v,2v+1): K = (v<4 ? 2v : 16+2(v-4)) + 8*half
__device__ __forceinline__ int a_frag_idx(int mt, int kt, int KT, int m15, int kk) {
    int half = (kk >> 3) & 1;
    int lane = half * 16 + m15;
    int v    = ((kk & 7) >> 1) + ((kk >> 4) << 2);
    int e    = kk & 1;
    return ((mt * KT + kt) * 32 + lane) * 16 + v * 2 + e;
}

// ---------------- weight pre-pack: B-fragment (32x16 KxN) layout ------------
// lanes 0-15 hold K in {0..7,16..23}, lanes 16-31 hold K in {8..15,24..31} (matches A-frag).
__global__ void pack_b_kernel(const float* __restrict__ W, _Float16* __restrict__ frag,
                              int K, int KT) {
    int flat = blockIdx.x * 256 + threadIdx.x;         // over H3*K elements
    int e    = flat & 15;
    int lane = (flat >> 4) & 31;
    int kt   = (flat >> 9) % KT;
    int nt   = (flat >> 9) / KT;
    int kk   = ((e >> 3) << 4) + ((lane >> 4) << 3) + (e & 7);
    int nn   = lane & 15;
    int n    = nt * 16 + nn;
    int k    = kt * 32 + kk;
    frag[flat] = (_Float16)W[(size_t)n * K + k];       // B[k][n] = W[n][k] (W row-major 3H x K)
}

__global__ void init_bar_kernel(int* bar) { if (threadIdx.x == 0) *bar = 0; }

// ---------------- persistent DeltaGRU scan kernel ---------------------------
__global__ __launch_bounds__(256)
void delta_gru_kernel(const float* __restrict__ x,
                      const float* __restrict__ bias,
                      float* __restrict__ out,
                      const _Float16* __restrict__ wihf,
                      const _Float16* __restrict__ whhf,
                      _Float16* __restrict__ dxf,    // [2][NBATCH*NI] halves, A-frag layout
                      _Float16* __restrict__ dhf,    // [2][NBATCH*NH] halves, A-frag layout
                      int* __restrict__ bar) {
    // recurrent state + accumulator exchange
    __shared__ float s_memx[3][NBATCH][16];
    __shared__ float s_memch[NBATCH][16];
    __shared__ float s_hid [NBATCH][16];
    __shared__ float s_hidp[NBATCH][16];
    __shared__ float s_inpp[NBATCH][16];
    __shared__ float s_acc [8][32][8];
    // resident weight B-fragments for this block's 3 gate column-tiles
    __shared__ __align__(16) _Float16 s_whh[3 * KT_H * 32 * 16];  // 48 KB
    __shared__ __align__(16) _Float16 s_wih[3 * KT_I * 32 * 16];  // 24 KB

    const int tid  = threadIdx.x;
    const int blk  = blockIdx.x;
    const int lane = tid & 31;
    const int w    = tid >> 5;

    // ---- one-time: stage this block's weight fragments into LDS ----
    // per gate g, global source is contiguous: whhf[nt*KT_H*512 ..], nt = g*32+blk
    for (int i = tid; i < 3 * KT_H * 32 * 2; i += 256) {           // 16B chunks
        int g   = i / (KT_H * 32 * 2);
        int rem = i % (KT_H * 32 * 2);
        int nt  = g * 32 + blk;
        ((float4*)s_whh)[i] = ((const float4*)(whhf + (size_t)nt * KT_H * 512))[rem];
    }
    for (int i = tid; i < 3 * KT_I * 32 * 2; i += 256) {
        int g   = i / (KT_I * 32 * 2);
        int rem = i % (KT_I * 32 * 2);
        int nt  = g * 32 + blk;
        ((float4*)s_wih)[i] = ((const float4*)(wihf + (size_t)nt * KT_I * 512))[rem];
    }
    // init recurrent state (hidden/prev = 0, mem_x = bias slice, mem_ch = 0)
    for (int ee = tid; ee < NBATCH * 16; ee += 256) {
        int m = ee >> 4, jj = ee & 15;
        s_memch[m][jj] = 0.f; s_hid[m][jj] = 0.f;
        s_hidp[m][jj] = 0.f;  s_inpp[m][jj] = 0.f;
        int base = 16 * blk + jj;
        s_memx[0][m][jj] = bias[base];
        s_memx[1][m][jj] = bias[NH + base];
        s_memx[2][m][jj] = bias[2 * NH + base];
    }
    __syncthreads();

    for (int t = 0; t < SEQ; ++t) {
        const int buf = t & 1;
        _Float16* dh = dhf + buf * (NBATCH * NH);
        _Float16* dx = dxf + buf * (NBATCH * NI);

        // ---- delta stage: thresholded deltas, written pre-swizzled to staging ----
        for (int ee = tid; ee < NBATCH * 16; ee += 256) {
            int m = ee >> 4, jj = ee & 15;
            float h = s_hid[m][jj], hp = s_hidp[m][jj];
            float d = h - hp;
            bool fire = fabsf(d) >= THRESH;
            float dv = fire ? d : 0.f;
            if (fire) s_hidp[m][jj] = h;
            int k = 16 * blk + jj;
            dh[a_frag_idx(m >> 4, k >> 5, KT_H, m & 15, k & 31)] = (_Float16)dv;
        }
        if (blk < 16) {  // first 16 blocks own the 256 input columns
            for (int ee = tid; ee < NBATCH * 16; ee += 256) {
                int m = ee >> 4, jj = ee & 15;
                int col = 16 * blk + jj;
                float xc = quantf(x[((size_t)t * NBATCH + m) * NI + col],
                                  256.f, 1.f / 256.f, 65536.f);
                float d = xc - s_inpp[m][jj];
                bool fire = fabsf(d) >= THRESH;
                float dv = fire ? d : 0.f;
                if (fire) s_inpp[m][jj] = xc;
                dx[a_frag_idx(m >> 4, col >> 5, KT_I, m & 15, col & 31)] = (_Float16)dv;
            }
            // prefetch next step's input slice into cache (global_prefetch_b8)
            if (t + 1 < SEQ && tid < NBATCH) {
                __builtin_prefetch(&x[((size_t)(t + 1) * NBATCH + tid) * NI + 16 * blk], 0, 0);
            }
        }

        // ---- one global barrier per step (monotonic counter, sense = t) ----
        __syncthreads();
        if (tid == 0) {
            __threadfence();
            atomicAdd(bar, 1);
            const int target = NBLK * (t + 1);
            while (atomicAdd(bar, 0) < target) __builtin_amdgcn_s_sleep(1);
            __threadfence();
        }
        __syncthreads();

        // ---- GEMM stage: 8 waves x (16x16 tile over full K), f16 WMMA ----
        // A fragments (deltas) from global/L2, B fragments (weights) from LDS.
        // Two interleaved accumulators break the WMMA D->C dependency chain.
        {
            const int g  = (w < 4) ? (w >> 1) : 2;   // 0=r,1=u,2=c
            const int mt = w & 1;                    // batch half
            const bool hasIH = (w < 6);              // r,u fused; w4/5 = c(ih)
            const bool hasHH = (w < 4) || (w >= 6);  // r,u fused; w6/7 = c(hh)
            v8f acc0 = {0.f,0.f,0.f,0.f,0.f,0.f,0.f,0.f};
            v8f acc1 = {0.f,0.f,0.f,0.f,0.f,0.f,0.f,0.f};
            if (hasIH) {
#pragma unroll
                for (int kt = 0; kt < KT_I; kt += 2) {
                    v16h a0 = *(const v16h*)(dx + ((mt * KT_I + kt) * 32 + lane) * 16);
                    v16h b0 = *(const v16h*)(&s_wih[((g * KT_I + kt) * 32 + lane) * 16]);
                    v16h a1 = *(const v16h*)(dx + ((mt * KT_I + kt + 1) * 32 + lane) * 16);
                    v16h b1 = *(const v16h*)(&s_wih[((g * KT_I + kt + 1) * 32 + lane) * 16]);
                    acc0 = __builtin_amdgcn_wmma_f32_16x16x32_f16(false, a0, false, b0,
                                                                  (short)0, acc0, false, false);
                    acc1 = __builtin_amdgcn_wmma_f32_16x16x32_f16(false, a1, false, b1,
                                                                  (short)0, acc1, false, false);
                }
            }
            if (hasHH) {
#pragma unroll
                for (int kt = 0; kt < KT_H; kt += 2) {
                    v16h a0 = *(const v16h*)(dh + ((mt * KT_H + kt) * 32 + lane) * 16);
                    v16h b0 = *(const v16h*)(&s_whh[((g * KT_H + kt) * 32 + lane) * 16]);
                    v16h a1 = *(const v16h*)(dh + ((mt * KT_H + kt + 1) * 32 + lane) * 16);
                    v16h b1 = *(const v16h*)(&s_whh[((g * KT_H + kt + 1) * 32 + lane) * 16]);
                    acc0 = __builtin_amdgcn_wmma_f32_16x16x32_f16(false, a0, false, b0,
                                                                  (short)0, acc0, false, false);
                    acc1 = __builtin_amdgcn_wmma_f32_16x16x32_f16(false, a1, false, b1,
                                                                  (short)0, acc1, false, false);
                }
            }
#pragma unroll
            for (int i = 0; i < 8; ++i) s_acc[w][lane][i] = acc0[i] + acc1[i];
        }
        __syncthreads();

        // ---- gate stage: quantized GRU update for this block's 16 units ----
        for (int ee = tid; ee < NBATCH * 16; ee += 256) {
            int m = ee >> 4, jj = ee & 15;
            int mt = m >> 4, mr = m & 15;
            int cl = (mr >> 3) * 16 + jj;   // C-layout: lane = (M/8)*16 + N
            int cv = mr & 7;                // C-layout: vgpr = M % 8
            float g_r  = s_acc[0 + mt][cl][cv];
            float g_u  = s_acc[2 + mt][cl][cv];
            float g_ci = s_acc[4 + mt][cl][cv];
            float g_ch = s_acc[6 + mt][cl][cv];

            float sum_r    = g_r  + s_memx[0][m][jj];   // ax_r + ah_r (mem_x folded in)
            float sum_u    = g_u  + s_memx[1][m][jj];
            float ax_c     = g_ci + s_memx[2][m][jj];
            float mem_ch_n = g_ch + s_memch[m][jj];

            float rr  = quantf(sigm(quantf(sum_r, 256.f, 1.f/256.f, 65536.f)), 16.f, 1.f/16.f, 4096.f);
            float uu  = quantf(sigm(quantf(sum_u, 256.f, 1.f/256.f, 65536.f)), 16.f, 1.f/16.f, 4096.f);
            float qch = quantf(mem_ch_n, 256.f, 1.f/256.f, 65536.f);
            float acc_c = ax_c + rr * qch;
            float cc  = quantf(tanhf(quantf(acc_c, 256.f, 1.f/256.f, 65536.f)), 16.f, 1.f/16.f, 4096.f);
            float aa  = quantf((1.f - uu) * cc, 256.f, 1.f/256.f, 65536.f);
            float bb  = quantf(uu * s_hid[m][jj], 256.f, 1.f/256.f, 65536.f);
            float nh  = quantf(aa + bb, 256.f, 1.f/256.f, 65536.f);

            s_memx[0][m][jj] = sum_r;
            s_memx[1][m][jj] = sum_u;
            s_memx[2][m][jj] = ax_c;
            s_memch[m][jj]   = mem_ch_n;
            s_hid[m][jj]     = nh;
            out[((size_t)t * NBATCH + m) * NH + 16 * blk + jj] = nh;
        }
        __syncthreads();
    }
}

// ---------------------------------------------------------------------------
extern "C" void kernel_launch(void* const* d_in, const int* in_sizes, int n_in,
                              void* d_out, int out_size, void* d_ws, size_t ws_size,
                              hipStream_t stream) {
    (void)in_sizes; (void)n_in; (void)out_size; (void)ws_size;
    const float* x    = (const float*)d_in[0];
    const float* wih  = (const float*)d_in[1];  // (3H, NI) row-major
    const float* whh  = (const float*)d_in[2];  // (3H, NH) row-major
    const float* bias = (const float*)d_in[3];  // (3H)
    float* out = (float*)d_out;

    char* ws = (char*)d_ws;
    size_t off = 0;
    auto carve = [&](size_t bytes) {
        void* p = ws + off;
        off = (off + bytes + 255) & ~(size_t)255;
        return p;
    };
    _Float16* wihf = (_Float16*)carve(sizeof(_Float16) * (size_t)H3 * NI);
    _Float16* whhf = (_Float16*)carve(sizeof(_Float16) * (size_t)H3 * NH);
    _Float16* dxf  = (_Float16*)carve(sizeof(_Float16) * 2 * NBATCH * NI);
    _Float16* dhf  = (_Float16*)carve(sizeof(_Float16) * 2 * NBATCH * NH);
    int*      bar  = (int*)carve(256);

    hipLaunchKernelGGL(init_bar_kernel, dim3(1), dim3(32), 0, stream, bar);
    hipLaunchKernelGGL(pack_b_kernel, dim3((H3 * NI) / 256), dim3(256), 0, stream,
                       wih, wihf, NI, KT_I);
    hipLaunchKernelGGL(pack_b_kernel, dim3((H3 * NH) / 256), dim3(256), 0, stream,
                       whh, whhf, NH, KT_H);
    hipLaunchKernelGGL(delta_gru_kernel, dim3(NBLK), dim3(256), 0, stream,
                       x, bias, out, wihf, whhf, dxf, dhf, bar);
}
